// KANLayer_50362786513073
// MI455X (gfx1250) — compile-verified, matrix operand order checked by reading, not run on hardware
//
#include <hip/hip_runtime.h>
#include <hip/hip_bf16.h>

// ---------------------------------------------------------------------------
// KAN layer on gfx1250:
//   out(1024x512) = A(1024x6144) @ B(6144x512), K-slots per input channel i:
//   [silu(x), basis_0..10(x)] vs [w[i,o], w[i,o]*c[i,o,k]].
// GEMM uses v_wmma_f32_16x16x32_f16 with double-buffered LDS tiles filled by
// the Tensor Data Mover (tensor_load_to_lds, TENSORcnt), split-K=4 + reduce.
// ---------------------------------------------------------------------------

typedef __attribute__((ext_vector_type(16))) _Float16 v16h;
typedef __attribute__((ext_vector_type(8)))  float    v8f;
typedef __attribute__((ext_vector_type(4)))  unsigned int u32x4;
typedef __attribute__((ext_vector_type(8)))  int          i32x8;
typedef __attribute__((ext_vector_type(4)))  int          i32x4;

#define BATCH   1024
#define N_IN    512
#define N_OUT   512
#define KB      11
#define KPC     12
#define KTOT    (N_IN * KPC)     // 6144

#define BM 128
#define BN 64
#define BK 32
#define LDA 40                   // LDS row stride in halves (80 B) via TDM pad
#define LDB 40
#define KSPLIT 4
#define KLEN   (KTOT / KSPLIT)   // 1536
#define NIT    (KLEN / BK)       // 48

union un16 { uint4 u[2]; v16h v; };

// --- TDM: async 2D tile load (f16 elements) global -> LDS ------------------
// Row length tile_d0 = 32 halves = 16 DWORDs; hardware pads +4 DWORDs per row
// (pad_interval=3 -> 16 DWORDs, pad_amount=3 -> 4 DWORDs) => 40-half stride.
__device__ __forceinline__ void tdm_load_tile(unsigned lds_addr,
                                              const _Float16* gptr,
                                              unsigned tile_d1 /*rows*/,
                                              unsigned stride0 /*elems*/) {
  unsigned long long ga = (unsigned long long)(uintptr_t)gptr;
  u32x4 g0;
  g0[0] = 1u;                                            // count=1, user mode
  g0[1] = lds_addr;                                      // LDS byte address
  g0[2] = (unsigned)(ga & 0xFFFFFFFFu);                  // global_addr[31:0]
  g0[3] = (unsigned)((ga >> 32) & 0x1FFFFFFu) | (2u << 30); // addr[56:32]|type=2
  i32x8 g1;
  g1[0] = (int)((1u << 16)        // data_size = 2 bytes
              | (1u << 20)        // pad_enable
              | (3u << 22)        // pad_interval: 16 DWORDs
              | (3u << 25));      // pad_amount:   4 DWORDs
  g1[1] = (int)(((unsigned)BK & 0xFFFFu) << 16);         // tensor_dim0 lo16
  g1[2] = (int)((tile_d1 & 0xFFFFu) << 16);              // td0 hi16=0 | tensor_dim1 lo16
  g1[3] = (int)((unsigned)BK << 16);                     // td1 hi16=0 | tile_dim0
  g1[4] = (int)(tile_d1 & 0xFFFFu);                      // tile_dim1 | tile_dim2=0
  g1[5] = (int)stride0;                                  // tensor_dim0_stride lo32
  g1[6] = 0;                                             // stride hi | dim1_stride lo
  g1[7] = 0;
  i32x4 g2 = {0, 0, 0, 0};
  i32x4 g3 = {0, 0, 0, 0};
  i32x8 g4 = {0, 0, 0, 0, 0, 0, 0, 0};                   // unused (VADDR4 slot)
  __builtin_amdgcn_tensor_load_to_lds(g0, g1, g2, g3, g4, 0);
}

// --- prep A: silu + dense cubic B-spline basis (uniform knots) -------------
__global__ __launch_bounds__(256) void kan_prep_a(const float* __restrict__ x,
                                                  _Float16* __restrict__ Af) {
  int idx = blockIdx.x * blockDim.x + threadIdx.x;       // b*N_IN + i
  if (idx >= BATCH * N_IN) return;
  float xv = x[idx];
  float s = xv / (1.0f + __expf(-xv));

  const float h = 0.75f, t0 = -5.25f;
  float B[14];
#pragma unroll
  for (int j = 0; j < 14; ++j) {
    float tj = t0 + h * (float)j;
    B[j] = (xv >= tj && xv < tj + h) ? 1.0f : 0.0f;
  }
#pragma unroll
  for (int d = 1; d <= 3; ++d) {
    float inv = 1.0f / (h * (float)d);
#pragma unroll 13
    for (int j = 0; j < 14 - d; ++j) {
      float tj = t0 + h * (float)j;
      B[j] = (xv - tj) * inv * B[j] + ((tj + h * (float)(d + 1)) - xv) * inv * B[j + 1];
    }
  }

  union { _Float16 hv[12]; uint2 u[3]; } pk;
  pk.hv[0] = (_Float16)s;
#pragma unroll
  for (int k = 0; k < KB; ++k) pk.hv[k + 1] = (_Float16)B[k];
  uint2* dst = (uint2*)Af + (size_t)idx * 3;
  dst[0] = pk.u[0]; dst[1] = pk.u[1]; dst[2] = pk.u[2];
}

// --- prep B: fuse w into c; store transposed (o-major, K-contiguous) -------
__global__ __launch_bounds__(256) void kan_prep_b(const float* __restrict__ c,
                                                  const float* __restrict__ w,
                                                  _Float16* __restrict__ Bt) {
  int idx = blockIdx.x * blockDim.x + threadIdx.x;       // i*N_OUT + o
  if (idx >= N_IN * N_OUT) return;
  int i = idx / N_OUT, o = idx % N_OUT;
  float wv = w[idx];
  const float* cp = c + (size_t)idx * KB;

  union { _Float16 hv[12]; uint2 u[3]; } pk;
  pk.hv[0] = (_Float16)wv;
#pragma unroll
  for (int k = 0; k < KB; ++k) pk.hv[k + 1] = (_Float16)(wv * cp[k]);
  uint2* dst = (uint2*)Bt + ((size_t)o * KTOT + (size_t)i * KPC) / 4;
  dst[0] = pk.u[0]; dst[1] = pk.u[1]; dst[2] = pk.u[2];
}

// --- GEMM: TDM double-buffered tiles + v_wmma_f32_16x16x32_f16, split-K ----
__global__ __launch_bounds__(256) void kan_gemm(const _Float16* __restrict__ A,
                                                const _Float16* __restrict__ Bt,
                                                float* __restrict__ part) {
  __shared__ _Float16 sA[2][BM * LDA];
  __shared__ _Float16 sB[2][BN * LDB];

  const int tid  = threadIdx.x;
  const int lane = tid & 31;
  const int wave = tid >> 5;      // 0..7
  const int wm   = wave & 3;      // 4 waves along M
  const int wn   = wave >> 2;     // 2 waves along N
  const int bn   = blockIdx.x;    // 0..7
  const int bm   = blockIdx.y;    // 0..7
  const int ks   = blockIdx.z;    // 0..3 split-K
  const int kbase = ks * KLEN;

  const _Float16* Ag = A  + (size_t)bm * BM * KTOT + kbase;
  const _Float16* Bg = Bt + (size_t)bn * BN * KTOT + kbase;

  const unsigned ldsA0 = (unsigned)(uintptr_t)&sA[0][0];
  const unsigned ldsA1 = (unsigned)(uintptr_t)&sA[1][0];
  const unsigned ldsB0 = (unsigned)(uintptr_t)&sB[0][0];
  const unsigned ldsB1 = (unsigned)(uintptr_t)&sB[1][0];

  v8f acc[2][2] = {};

  const int kc = (lane >> 4) * 16;    // frag K sub-chunk (contiguous k = 16h + s, same for A and B)
  const int rl = lane & 15;

  // prologue: wave 0 kicks off tile 0 into buffer 0
  if (wave == 0) {
    tdm_load_tile(ldsA0, Ag, BM, KTOT);
    tdm_load_tile(ldsB0, Bg, BN, KTOT);
  }

  for (int it = 0; it < NIT; ++it) {
    const int cur = it & 1;
    const _Float16* sAc = sA[cur];
    const _Float16* sBc = sB[cur];

    if (wave == 0) {
      if (it + 1 < NIT) {
        const int koff = (it + 1) * BK;
        tdm_load_tile(cur ? ldsA0 : ldsA1, Ag + koff, BM, KTOT);
        tdm_load_tile(cur ? ldsB0 : ldsB1, Bg + koff, BN, KTOT);
        __builtin_amdgcn_s_wait_tensorcnt(2);   // current buffer's pair done
      } else {
        __builtin_amdgcn_s_wait_tensorcnt(0);
      }
    }
    __syncthreads();                            // tile[cur] visible to all

    v16h af[2], bf[2];
#pragma unroll
    for (int tm = 0; tm < 2; ++tm) {
      const _Float16* p = &sAc[(wm * 32 + tm * 16 + rl) * LDA + kc];
      un16 t; t.u[0] = *(const uint4*)p; t.u[1] = *(const uint4*)(p + 8);
      af[tm] = t.v;
    }
#pragma unroll
    for (int tn = 0; tn < 2; ++tn) {
      const _Float16* p = &sBc[(wn * 32 + tn * 16 + rl) * LDB + kc];
      un16 t; t.u[0] = *(const uint4*)p; t.u[1] = *(const uint4*)(p + 8);
      bf[tn] = t.v;
    }
#pragma unroll
    for (int tm = 0; tm < 2; ++tm)
#pragma unroll
      for (int tn = 0; tn < 2; ++tn)
        acc[tm][tn] = __builtin_amdgcn_wmma_f32_16x16x32_f16(
            false, af[tm], false, bf[tn], (short)0, acc[tm][tn], false, false);
    __syncthreads();                            // done reading tile[cur]
  }

  float* outp = part + (size_t)ks * BATCH * N_OUT;
  const int mw = bm * BM + wm * 32;
  const int nw = bn * BN + wn * 32;
  const int hh = (lane >> 4) * 8;
#pragma unroll
  for (int tm = 0; tm < 2; ++tm)
#pragma unroll
    for (int tn = 0; tn < 2; ++tn) {
      int n     = nw + tn * 16 + rl;
      int mbase = mw + tm * 16 + hh;
#pragma unroll
      for (int r = 0; r < 8; ++r)
        outp[(size_t)(mbase + r) * N_OUT + n] = acc[tm][tn][r];
    }
}

// --- deterministic split-K reduction ---------------------------------------
__global__ __launch_bounds__(256) void kan_reduce(const float4* __restrict__ p,
                                                  float4* __restrict__ out) {
  const int Q = (BATCH * N_OUT) / 4;
  int i = blockIdx.x * blockDim.x + threadIdx.x;
  if (i >= Q) return;
  float4 a = p[i], b = p[i + Q], c = p[i + 2 * Q], d = p[i + 3 * Q];
  float4 r;
  r.x = (a.x + b.x) + (c.x + d.x);
  r.y = (a.y + b.y) + (c.y + d.y);
  r.z = (a.z + b.z) + (c.z + d.z);
  r.w = (a.w + b.w) + (c.w + d.w);
  out[i] = r;
}

extern "C" void kernel_launch(void* const* d_in, const int* in_sizes, int n_in,
                              void* d_out, int out_size, void* d_ws, size_t ws_size,
                              hipStream_t stream) {
  const float* x = (const float*)d_in[0];
  const float* c = (const float*)d_in[1];
  const float* w = (const float*)d_in[2];

  char* ws = (char*)d_ws;
  _Float16* Af   = (_Float16*)ws;                                    // 12 MB
  _Float16* Btp  = (_Float16*)(ws + (size_t)BATCH * KTOT * 2);       //  6 MB
  float*    part = (float*)(ws + (size_t)BATCH * KTOT * 2
                               + (size_t)N_OUT * KTOT * 2);          //  8 MB

  kan_prep_a<<<(BATCH * N_IN) / 256, 256, 0, stream>>>(x, Af);
  kan_prep_b<<<(N_IN * N_OUT) / 256, 256, 0, stream>>>(c, w, Btp);

  dim3 grid(N_OUT / BN, BATCH / BM, KSPLIT);   // (8, 8, 4) = 256 blocks
  kan_gemm<<<grid, 256, 0, stream>>>(Af, Btp, part);

  kan_reduce<<<(BATCH * N_OUT / 4) / 256, 256, 0, stream>>>(
      (const float4*)part, (float4*)d_out);
}
